// Last_Prop_34437047779570
// MI455X (gfx1250) — compile-verified, matrix operand order checked by reading, not run on hardware
//
#include <hip/hip_runtime.h>

// ---------------------------------------------------------------------------
// MI455X (gfx1250) implementation of the hypothesis-propagation block.
//  - f16 activations, channels-last [b][h][w][32] in workspace
//  - 3x3 convs as implicit GEMM with v_wmma_f32_16x16x32_f16 (K = 32 channels)
//  - 3-row LDS ring buffer per workgroup (144 KB dynamic LDS, CDNA5 320KB WGP)
//  - weights prepacked into ISA A-fragment layout (2x b128 loads per lane)
//  - lrelu as max(x, 0.2x): 2 VALU ops, no v_cmp/v_cndmask
// ---------------------------------------------------------------------------

typedef __attribute__((ext_vector_type(16))) _Float16 v16h;
typedef __attribute__((ext_vector_type(8)))  float    v8f;

#define BB   4
#define CH   32
#define HC   16
#define HH   384
#define WW   768
#define BAND 6
#define STRIPS (WW / 16)          // 48 strips of 16 pixels per row
#define ROWB  ((WW + 2) * 64)     // LDS bytes per padded row: 770 px * 32ch * 2B
#define SLOPE 0.2f

union V16 { uint4 q[2]; v16h h; };
union V8  { uint4 q;    _Float16 h[8]; };

__device__ __forceinline__ int iclamp(int v, int lo, int hi) {
  return v < lo ? lo : (v > hi ? hi : v);
}

// leaky-relu with slope < 1:  lrelu(x) = max(x, slope*x)  (exact)
__device__ __forceinline__ float lrelu(float x) {
  return fmaxf(x, SLOPE * x);
}

// ---------------------------------------------------------------------------
// Weight prepack: OIHW f32 -> f16 WMMA A-fragments.
// dst layout: [tap(9)][group(ocg)][lane(32)][elem(16)] f16
// lane L holds M = L&15 (out channel within group), element e holds
// K = 8*(L>>4) + (e<8 ? e : e+8)  (per CDNA5 ISA 16-bit A-matrix layout).
// ---------------------------------------------------------------------------
__global__ __launch_bounds__(256) void prepack_weights(
    const float* __restrict__ w, _Float16* __restrict__ dst, int ocg)
{
  int total = 9 * ocg * 32 * 16;
  for (int i = blockIdx.x * 256 + threadIdx.x; i < total;
       i += gridDim.x * 256) {
    int e    = i & 15;
    int frag = i >> 4;
    int lane = frag & 31;
    int gg   = (frag >> 5) % ocg;
    int tap  = frag / (32 * ocg);
    int oc   = gg * 16 + (lane & 15);
    int k    = 8 * (lane >> 4) + (e < 8 ? e : e + 8);
    int ty   = tap / 3, tx = tap % 3;
    dst[i] = (_Float16)w[((oc * CH + k) * 3 + ty) * 3 + tx];
  }
}

// ---------------------------------------------------------------------------
// Kernel 1: augmented hypothesis (bilinear warp cost volume, 4ch) fused with
// the 1x1 conv 20->32 + leaky-relu.  Output: f16 channels-last activation.
// ---------------------------------------------------------------------------
__global__ __launch_bounds__(256) void augment_conv1_kernel(
    const float* __restrict__ hyp,  const float* __restrict__ fea_l,
    const float* __restrict__ fea_r, const float* __restrict__ w_cn,
    const float* __restrict__ b_cn,  _Float16* __restrict__ out)
{
  int gid = blockIdx.x * 256 + threadIdx.x;
  int x = gid % WW;
  int y = (gid / WW) % HH;
  int b = gid / (WW * HH);
  const size_t HWp = (size_t)HH * WW;

  float disp = hyp[(size_t)b * HC * HWp + (size_t)y * WW + x];

  float la[CH];
  float v[4 + HC];
  float s0 = 0.f;
  const float* flb = fea_l + (size_t)b * CH * HWp + (size_t)y * WW + x;
  #pragma unroll
  for (int c = 0; c < CH; c++) { la[c] = flb[c * HWp]; s0 += fabsf(la[c]); }
  v[0] = s0;

  // y geometry (same for all three offsets)
  const float yscale = (float)HH / (float)(HH - 1);
  float y_pix = (float)y * yscale - 0.5f;
  float y0f = floorf(y_pix);
  float wy  = y_pix - y0f;
  int   y0  = (int)y0f, y1 = y0 + 1;
  float m_y0 = (y0 >= 0 && y0 < HH) ? 1.f : 0.f;
  float m_y1 = (y1 >= 0 && y1 < HH) ? 1.f : 0.f;
  int y0c = iclamp(y0, 0, HH - 1), y1c = iclamp(y1, 0, HH - 1);

  const float* frb = fea_r + (size_t)b * CH * HWp;
  const float xscale = (float)WW / (float)(WW - 1);
  #pragma unroll
  for (int o = 0; o < 3; o++) {
    float xd = fminf(fmaxf((float)x - disp + (float)(o - 1), 0.f),
                     (float)(WW - 1));
    float x_pix = xd * xscale - 0.5f;
    float x0f = floorf(x_pix);
    float wx  = x_pix - x0f;
    int   x0  = (int)x0f, x1 = x0 + 1;
    float m_x0 = (x0 >= 0 && x0 < WW) ? 1.f : 0.f;
    float m_x1 = (x1 >= 0 && x1 < WW) ? 1.f : 0.f;
    int x0c = iclamp(x0, 0, WW - 1), x1c = iclamp(x1, 0, WW - 1);
    float w00 = (1.f - wy) * (1.f - wx) * m_y0 * m_x0;
    float w01 = (1.f - wy) * wx         * m_y0 * m_x1;
    float w10 = wy         * (1.f - wx) * m_y1 * m_x0;
    float w11 = wy         * wx         * m_y1 * m_x1;
    size_t i00 = (size_t)y0c * WW + x0c;
    size_t i01 = (size_t)y0c * WW + x1c;
    size_t i10 = (size_t)y1c * WW + x0c;
    size_t i11 = (size_t)y1c * WW + x1c;
    float acc = 0.f;
    for (int c = 0; c < CH; c++) {
      const float* f = frb + (size_t)c * HWp;
      float warp = w00 * f[i00] + w01 * f[i01] + w10 * f[i10] + w11 * f[i11];
      acc += fabsf(la[c] - warp);
    }
    v[1 + o] = acc;
  }

  const float* hb = hyp + (size_t)b * HC * HWp + (size_t)y * WW + x;
  #pragma unroll
  for (int j = 0; j < HC; j++) v[4 + j] = hb[(size_t)j * HWp];

  size_t pix = (size_t)(b * HH + y) * WW + x;
  _Float16* op = out + pix * CH;
  #pragma unroll
  for (int oc = 0; oc < CH; oc++) {
    float s = b_cn[oc];
    #pragma unroll
    for (int j = 0; j < 20; j++) s += w_cn[oc * 20 + j] * v[j];
    op[oc] = (_Float16)lrelu(s);
  }
}

// ---------------------------------------------------------------------------
// Stage one padded activation row (770 px x 32ch f16 = 48.1 KB) into LDS.
// Pad pixels / out-of-image rows are written as zeros (conv zero padding).
// ---------------------------------------------------------------------------
__device__ __forceinline__ void loadRowToLds(
    char* dst, const _Float16* __restrict__ in, int b, int hh, int tid)
{
  const uint4* src = nullptr;
  if (hh >= 0 && hh < HH)
    src = (const uint4*)(in + (size_t)(b * HH + hh) * WW * CH);
  uint4* d = (uint4*)dst;
  for (int i = tid; i < (WW + 2) * 4; i += 256) {
    uint4 val = make_uint4(0u, 0u, 0u, 0u);
    int gx = (i >> 2) - 1;
    if (src != nullptr && gx >= 0 && gx < WW)
      val = src[(size_t)gx * 4 + (i & 3)];
    d[i] = val;
  }
}

// ---------------------------------------------------------------------------
// 3x3 conv, 32 in-channels, OCG*16 out-channels, implicit GEMM via WMMA.
// Workgroup = 8 waves, sweeps a BAND-row band with a 3-row LDS ring buffer.
// Each wave: fixed out-channel group g, strips s = sw, sw+8/OCG, ...
// Per 16-pixel strip: 9 taps -> 9x v_wmma_f32_16x16x32_f16 (K = 32 channels).
// ---------------------------------------------------------------------------
template <int OCG, bool RES, bool FINAL>
__global__ __launch_bounds__(256) void conv3x3_wmma_kernel(
    const _Float16* __restrict__ in,  const _Float16* __restrict__ wp,
    const float* __restrict__ bias,   const _Float16* __restrict__ res,
    _Float16* __restrict__ out16,     const float* __restrict__ hyp,
    float* __restrict__ outf)
{
  extern __shared__ char sm[];
  const int tid  = threadIdx.x;
  const int lane = tid & 31;
  const int wave = tid >> 5;
  const int hi   = lane >> 4;          // which half-wave (C/D row block)

  const int band = blockIdx.x % (HH / BAND);
  const int b    = blockIdx.x / (HH / BAND);
  const int h0   = band * BAND;

  const int g  = wave & (OCG - 1);     // out-channel group for this wave
  const int sw = wave / OCG;           // strip phase
  const int NW = 8 / OCG;

  // A-fragments (weights) resident in VGPRs for the whole band: 9 taps x 8 VGPRs
  V16 aw[9];
  #pragma unroll
  for (int t = 0; t < 9; t++) {
    const uint4* wq = (const uint4*)(wp + (size_t)((t * OCG + g) * 32 + lane) * 16);
    aw[t].q[0] = wq[0];
    aw[t].q[1] = wq[1];
  }
  float bias_r[8];
  #pragma unroll
  for (int r = 0; r < 8; r++) bias_r[r] = bias[g * 16 + hi * 8 + r];

  // prime ring buffer with rows h0-1 and h0
  loadRowToLds(sm + 0 * ROWB, in, b, h0 - 1, tid);
  loadRowToLds(sm + 1 * ROWB, in, b, h0,     tid);

  for (int r = 0; r < BAND; r++) {
    const int h = h0 + r;
    loadRowToLds(sm + ((r + 2) % 3) * ROWB, in, b, h + 1, tid);
    __syncthreads();

    // prefetch next band row while we compute (global_prefetch_b8)
    if (h + 2 < HH) {
      const char* pf = (const char*)(in + (size_t)(b * HH + h + 2) * WW * CH);
      __builtin_prefetch(pf + tid * 192, 0, 3);
    }

    const char* rbs[3] = { sm + (r % 3) * ROWB,
                           sm + ((r + 1) % 3) * ROWB,
                           sm + ((r + 2) % 3) * ROWB };

    for (int s = sw; s < STRIPS; s += NW) {       // wave-uniform loop
      v8f acc;
      #pragma unroll
      for (int rr = 0; rr < 8; rr++) acc[rr] = bias_r[rr];

      #pragma unroll
      for (int t = 0; t < 9; t++) {
        const int dy = t / 3;            // 0..2 -> rows h-1,h,h+1
        const int dx = t % 3 - 1;        // -1..1
        const int p  = s * 16 + (lane & 15) + dx;        // -1..768
        const char* base = rbs[dy] + (size_t)(p + 1) * 64 + hi * 16;
        V16 bf;                                          // B: K=32 x N=16
        bf.q[0] = *(const uint4*)(base);                 // ds_load_b128
        bf.q[1] = *(const uint4*)(base + 32);            // ds_load_b128
        acc = __builtin_amdgcn_wmma_f32_16x16x32_f16(
                  false, aw[t].h, false, bf.h, (short)0, acc, false, false);
      }

      const int p0 = s * 16 + (lane & 15);
      if (FINAL) {
        // D = conv + hyp, relu on channel 0, f32 NCHW store to d_out
        #pragma unroll
        for (int rr = 0; rr < 8; rr++) {
          int oc = hi * 8 + rr;
          size_t idx = ((size_t)(b * HC + oc) * HH + h) * WW + p0;
          float val = acc[rr] + hyp[idx];
          val = (oc == 0) ? fmaxf(val, 0.f) : val;
          outf[idx] = val;
        }
      } else {
        size_t pix = (size_t)(b * HH + h) * WW + p0;
        const int cb = g * 16 + hi * 8;   // 8 consecutive channels per lane
        if (RES) {
          V8 rf; rf.q = *(const uint4*)(res + pix * CH + cb);
          #pragma unroll
          for (int rr = 0; rr < 8; rr++) acc[rr] += (float)rf.h[rr];
        }
        V8 of;
        #pragma unroll
        for (int rr = 0; rr < 8; rr++)
          of.h[rr] = (_Float16)lrelu(acc[rr]);          // v_mul + v_max
        *(uint4*)(out16 + pix * CH + cb) = of.q;        // b128 store
      }
    }
    __syncthreads();
  }
}

// ---------------------------------------------------------------------------
extern "C" void kernel_launch(void* const* d_in, const int* in_sizes, int n_in,
                              void* d_out, int out_size, void* d_ws,
                              size_t ws_size, hipStream_t stream)
{
  (void)in_sizes; (void)n_in; (void)out_size; (void)ws_size;

  const float* hyp   = (const float*)d_in[0];
  const float* fea_l = (const float*)d_in[1];
  const float* fea_r = (const float*)d_in[2];
  const float* w_cn  = (const float*)d_in[3];
  const float* b_cn  = (const float*)d_in[4];
  const float* r0_w1 = (const float*)d_in[5];
  const float* r0_b1 = (const float*)d_in[6];
  const float* r0_w2 = (const float*)d_in[7];
  const float* r0_b2 = (const float*)d_in[8];
  const float* r1_w1 = (const float*)d_in[9];
  const float* r1_b1 = (const float*)d_in[10];
  const float* r1_w2 = (const float*)d_in[11];
  const float* r1_b2 = (const float*)d_in[12];
  const float* cn_w  = (const float*)d_in[13];
  const float* cn_b  = (const float*)d_in[14];
  float* outp = (float*)d_out;

  // workspace layout: [weight packs | actA | actB | actC]
  char* ws = (char*)d_ws;
  const size_t WPL    = (size_t)9 * 2 * 32 * 16;                 // f16 per layer
  const size_t ABYTES = (size_t)BB * HH * WW * CH * sizeof(_Float16);
  _Float16* wpack = (_Float16*)ws;
  _Float16* actA  = (_Float16*)(ws + 128 * 1024);
  _Float16* actB  = (_Float16*)(ws + 128 * 1024 + ABYTES);
  _Float16* actC  = (_Float16*)(ws + 128 * 1024 + 2 * ABYTES);

  // 1) prepack conv weights into WMMA A-fragment layout
  prepack_weights<<<dim3(36), 256, 0, stream>>>(r0_w1, wpack + 0 * WPL, 2);
  prepack_weights<<<dim3(36), 256, 0, stream>>>(r0_w2, wpack + 1 * WPL, 2);
  prepack_weights<<<dim3(36), 256, 0, stream>>>(r1_w1, wpack + 2 * WPL, 2);
  prepack_weights<<<dim3(36), 256, 0, stream>>>(r1_w2, wpack + 3 * WPL, 2);
  prepack_weights<<<dim3(18), 256, 0, stream>>>(cn_w,  wpack + 4 * WPL, 1);

  // 2) warp cost volume + fused 1x1 conv -> x0 (actA)
  augment_conv1_kernel<<<dim3((BB * HH * WW) / 256), 256, 0, stream>>>(
      hyp, fea_l, fea_r, w_cn, b_cn, actA);

  // 3) residual tower, WMMA conv layers (3-row LDS ring buffer ~144KB)
  const size_t lds  = (size_t)3 * ROWB;
  const dim3   grid(BB * (HH / BAND));
  conv3x3_wmma_kernel<2, false, false><<<grid, 256, lds, stream>>>(
      actA, wpack + 0 * WPL, r0_b1, nullptr, actB, nullptr, nullptr);
  conv3x3_wmma_kernel<2, true,  false><<<grid, 256, lds, stream>>>(
      actB, wpack + 1 * WPL, r0_b2, actA,    actC, nullptr, nullptr);
  conv3x3_wmma_kernel<2, false, false><<<grid, 256, lds, stream>>>(
      actC, wpack + 2 * WPL, r1_b1, nullptr, actB, nullptr, nullptr);
  conv3x3_wmma_kernel<2, true,  false><<<grid, 256, lds, stream>>>(
      actB, wpack + 3 * WPL, r1_b2, actC,    actA, nullptr, nullptr);

  // 4) final 3x3 conv 32->16, fused +hyp, relu(ch0), f32 NCHW output
  conv3x3_wmma_kernel<1, false, true><<<grid, 256, lds, stream>>>(
      actA, wpack + 4 * WPL, cn_b, nullptr, nullptr, hyp, outp);
}
